// SlotAttention_84920093376896
// MI455X (gfx1250) — compile-verified, matrix operand order checked by reading, not run on hardware
//
#include <hip/hip_runtime.h>
#include <stdint.h>

// ---------- types ----------
typedef __attribute__((ext_vector_type(16))) __bf16 v16bf;
typedef __attribute__((ext_vector_type(8)))  __bf16 v8bf;
typedef __attribute__((ext_vector_type(8)))  float  v8f;

#define WMMA_BF16(A,B,C) \
  __builtin_amdgcn_wmma_f32_16x16x32_bf16(false,(A),false,(B),(short)0,(C),false,false)

__device__ __forceinline__ v16bf cat8(v8bf lo, v8bf hi) {
  return __builtin_shufflevector(lo, hi, 0,1,2,3,4,5,6,7,8,9,10,11,12,13,14,15);
}

#define NB      64
#define NTOK    4096
#define HDIM    128
#define NS      7
#define SD      32
#define NSPLIT  8
#define SCALEF  0.17677669529663688f   // 32^-0.5
#define EPSA    1e-8f

// fragment K-index map for 16-bit A/B layouts:
// lane half h, element i (0..15)  ->  K = 8h + (i&7) + 16*(i>>3)
// inverse: h=(K>>3)&1, i=(K&7)+8*(K>>4)

// =====================================================================
// Kernel 1: fused input-LayerNorm + K/V projection (bf16 WMMA, f32 acc)
// grid (32, 64): 128 tokens per block of one batch. 256 threads = 8 waves.
// =====================================================================
__global__ __launch_bounds__(256) void k_lnkv(
    const float* __restrict__ x, const float* __restrict__ g, const float* __restrict__ bln,
    const float* __restrict__ Wk, const float* __restrict__ Wv,
    __bf16* __restrict__ kbuf, __bf16* __restrict__ vT)
{
  __shared__ float  part[128][2][2];
  __shared__ __bf16 xfrag[8][4][32][16];   // [row-tile][K-step][lane][elem]
  __shared__ __bf16 wfrag[4][4][32][16];   // [col-tile: k0,k1,v0,v1][K-step][lane][elem]

  const int b    = blockIdx.y;
  const int tok0 = blockIdx.x * 128;
  const int tid  = threadIdx.x;
  const int r    = tid >> 1, hs = tid & 1;

  // ---- LN stats: 2 threads per row, 64 contiguous floats each ----
  const float* xrow = x + ((size_t)b*NTOK + tok0 + r)*HDIM + hs*64;
  float xv[64];
  float sm = 0.f, sq = 0.f;
#pragma unroll
  for (int i = 0; i < 16; ++i) {
    float4 t = ((const float4*)xrow)[i];
    xv[4*i+0]=t.x; xv[4*i+1]=t.y; xv[4*i+2]=t.z; xv[4*i+3]=t.w;
    sm += t.x + t.y + t.z + t.w;
    sq += t.x*t.x + t.y*t.y + t.z*t.z + t.w*t.w;
  }
  part[r][hs][0] = sm; part[r][hs][1] = sq;

  // ---- stage Wk/Wv as B-fragments (bf16) ----
  for (int t2 = tid; t2 < 512; t2 += 256) {
    int ct = t2 >> 7, rem = t2 & 127, ks = rem >> 5, ln = rem & 31;
    int h = ln >> 4, n = ln & 15;
    const float* W = ((ct < 2) ? Wk : Wv) + (size_t)(16*(ct & 1) + n)*HDIM + ks*32;
#pragma unroll
    for (int u = 0; u < 8; ++u) {
      wfrag[ct][ks][ln][u]     = (__bf16)W[8*h + u];
      wfrag[ct][ks][ln][8 + u] = (__bf16)W[16 + 8*h + u];
    }
  }
  __syncthreads();

  // ---- finish LN, write x as bf16 A-fragments ----
  {
    float mean = (part[r][0][0] + part[r][1][0]) * (1.f/128.f);
    float var  = (part[r][0][1] + part[r][1][1]) * (1.f/128.f) - mean*mean;
    float rstd = rsqrtf(var + 1e-5f);
    const int rt = r >> 4, m = r & 15;
#pragma unroll
    for (int j = 0; j < 64; ++j) {
      int f = hs*64 + j;
      float y = (xv[j] - mean) * rstd * g[f] + bln[f];
      int ks = f >> 5, Kf = f & 31;
      int h = (Kf >> 3) & 1, ii = (Kf & 7) + 8*(Kf >> 4);
      xfrag[rt][ks][m + 16*h][ii] = (__bf16)y;
    }
  }
  __syncthreads();

  // ---- WMMA: each wave owns one 16-token row tile, all 4 col tiles ----
  const int w = tid >> 5, lane = tid & 31;
  const int n = lane & 15, h = lane >> 4;
  v16bf A[4];
#pragma unroll
  for (int ks = 0; ks < 4; ++ks) {
    v8bf lo = *(const v8bf*)&xfrag[w][ks][lane][0];
    v8bf hi = *(const v8bf*)&xfrag[w][ks][lane][8];
    A[ks] = cat8(lo, hi);
  }
#pragma unroll
  for (int ct = 0; ct < 4; ++ct) {
    v8f c = {0.f,0.f,0.f,0.f,0.f,0.f,0.f,0.f};
#pragma unroll
    for (int ks = 0; ks < 4; ++ks) {
      v8bf lo = *(const v8bf*)&wfrag[ct][ks][lane][0];
      v8bf hi = *(const v8bf*)&wfrag[ct][ks][lane][8];
      c = WMMA_BF16(A[ks], cat8(lo, hi), c);
    }
    if (ct < 2) {                 // k: row-major [B][N][32] bf16
      int col = 16*ct + n;
#pragma unroll
      for (int rr = 0; rr < 8; ++rr) {
        int tok = tok0 + 16*w + rr + 8*h;
        kbuf[((size_t)b*NTOK + tok)*SD + col] = (__bf16)c[rr];
      }
    } else {                      // v: transposed [B][32][N] bf16, vectorized store
      int col = 16*(ct - 2) + n;
      v8bf pk;
#pragma unroll
      for (int rr = 0; rr < 8; ++rr) pk[rr] = (__bf16)c[rr];
      *(v8bf*)&vT[((size_t)b*SD + col)*NTOK + tok0 + 16*w + 8*h] = pk;
    }
  }
}

// =====================================================================
// Kernel 2: one attention iteration (q, dots-WMMA, softmax-over-slots,
// raw-attn store, attn@v WMMA). grid (8 splits, 64 batches), 256 thr.
// =====================================================================
__global__ __launch_bounds__(256) void k_attn(
    const float* __restrict__ slots, const float* __restrict__ Wq,
    const float* __restrict__ gs, const float* __restrict__ bsb,
    const __bf16* __restrict__ kbuf, const __bf16* __restrict__ vT,
    float* __restrict__ attn_out, float* __restrict__ upd_part, float* __restrict__ row_part)
{
  __shared__ float  sl[7][32];
  __shared__ float  st[7][2];
  __shared__ __bf16 qfrag[32][16];
  __shared__ __bf16 ua[8][32][16];       // per-wave attn A-fragment staging
  __shared__ float  rowred[8][7][32];
  __shared__ float  updred[8][7][32];

  const int b = blockIdx.y, split = blockIdx.x;
  const int tid  = threadIdx.x;
  const int w    = tid >> 5, lane = tid & 31;
  const int n    = lane & 15, h = lane >> 4;

  // zero fragment scratch (pad rows 7..15 of A matrices stay zero)
  ((uint32_t*)qfrag)[tid] = 0u;
  for (int i = tid; i < 2048; i += 256) ((uint32_t*)ua)[i] = 0u;
  if (tid < 224) sl[tid >> 5][tid & 31] = slots[((size_t)b*NS + (tid >> 5))*SD + (tid & 31)];
  __syncthreads();

  if (tid < 7) {                          // LN stats of slots
    float s0 = 0.f, s1 = 0.f;
    for (int j = 0; j < 32; ++j) { float v = sl[tid][j]; s0 += v; s1 += v*v; }
    float mn = s0 * (1.f/32.f);
    st[tid][0] = mn;
    st[tid][1] = rsqrtf(s1*(1.f/32.f) - mn*mn + 1e-5f);
  }
  __syncthreads();
  if (tid < 224) {                        // q = LN(slots) @ Wq.T  -> A-fragment
    int s = tid >> 5, o = tid & 31;
    float mn = st[s][0], rstd = st[s][1];
    float acc = 0.f;
    for (int j = 0; j < 32; ++j)
      acc += ((sl[s][j] - mn)*rstd*gs[j] + bsb[j]) * Wq[o*32 + j];
    int hh = (o >> 3) & 1, ii = (o & 7) + 8*(o >> 4);
    qfrag[s + 16*hh][ii] = (__bf16)acc;
  }
  __syncthreads();

  v16bf aq;
  { v8bf lo = *(const v8bf*)&qfrag[lane][0];
    v8bf hi = *(const v8bf*)&qfrag[lane][8];
    aq = cat8(lo, hi); }

  const __bf16* kb = kbuf + (size_t)b*NTOK*SD;
  const __bf16* vb = vT   + (size_t)b*SD*NTOK;
  float* ao = attn_out + (size_t)b*NS*NTOK;

  float acc7[7] = {0.f,0.f,0.f,0.f,0.f,0.f,0.f};
  v8f cu0 = {0.f,0.f,0.f,0.f,0.f,0.f,0.f,0.f};
  v8f cu1 = cu0;

  const int tokW = split*512 + 64*w;      // 64 tokens per wave, 2 chunks of 32
  for (int cch = 0; cch < 2; ++cch) {
    const int cb = tokW + 32*cch;
#pragma unroll
    for (int tt = 0; tt < 2; ++tt) {
      const int tb = cb + 16*tt;
      const __bf16* kr = kb + (size_t)(tb + n)*SD;
      v8bf klo = *(const v8bf*)(kr + 8*h);
      v8bf khi = *(const v8bf*)(kr + 16 + 8*h);
      v8f cd = {0.f,0.f,0.f,0.f,0.f,0.f,0.f,0.f};
      cd = WMMA_BF16(aq, cat8(klo, khi), cd);   // dots tile [16 slots x 16 tok]
      if (lane < 16) {                          // softmax over the 7 slots (lane-local)
        float a7[7];
#pragma unroll
        for (int s = 0; s < 7; ++s) a7[s] = cd[s]*SCALEF;
        float mx = a7[0];
#pragma unroll
        for (int s = 1; s < 7; ++s) mx = fmaxf(mx, a7[s]);
        float ssum = 0.f;
#pragma unroll
        for (int s = 0; s < 7; ++s) { a7[s] = __expf(a7[s]-mx); ssum += a7[s]; }
        float inv = 1.f/ssum;
        const int tok = tb + n;
        const int hh = (n >> 3) & 1;
#pragma unroll
        for (int s = 0; s < 7; ++s) {
          float av = a7[s]*inv + EPSA;
          ao[(size_t)s*NTOK + tok] = av;        // raw attn (normalized later)
          acc7[s] += av;
          ua[w][s + 16*hh][(n & 7) + 8*tt] = (__bf16)av;
        }
      }
    }
    // updates += attn_chunk[16x32] @ v_chunk[32x32]   (2 col tiles)
    v16bf au;
    { v8bf lo = *(const v8bf*)&ua[w][lane][0];
      v8bf hi = *(const v8bf*)&ua[w][lane][8];
      au = cat8(lo, hi); }
    {
      const __bf16* vr0 = vb + (size_t)(n)*NTOK + cb;
      v8bf vlo = *(const v8bf*)(vr0 + 8*h);
      v8bf vhi = *(const v8bf*)(vr0 + 16 + 8*h);
      cu0 = WMMA_BF16(au, cat8(vlo, vhi), cu0);
    }
    {
      const __bf16* vr1 = vb + (size_t)(16 + n)*NTOK + cb;
      v8bf vlo = *(const v8bf*)(vr1 + 8*h);
      v8bf vhi = *(const v8bf*)(vr1 + 16 + 8*h);
      cu1 = WMMA_BF16(au, cat8(vlo, vhi), cu1);
    }
  }

  // ---- cross-wave reduction of rowsums and update partials ----
#pragma unroll
  for (int s = 0; s < 7; ++s) rowred[w][s][lane] = acc7[s];   // lanes>=16 are zero
  if (lane < 16) {
#pragma unroll
    for (int s = 0; s < 7; ++s) {
      updred[w][s][n]      = cu0[s];
      updred[w][s][16 + n] = cu1[s];
    }
  }
  __syncthreads();
  if (tid < 224) {
    int s = tid >> 5, l = tid & 31;
    float rv = 0.f, uv = 0.f;
#pragma unroll
    for (int wv = 0; wv < 8; ++wv) { rv += rowred[wv][s][l]; uv += updred[wv][s][l]; }
    upd_part[(((size_t)split*NB + b)*NS + s)*SD + l] = uv;
    for (int off = 16; off > 0; off >>= 1) rv += __shfl_down(rv, off, 32);
    if (l == 0) row_part[((size_t)split*NB + b)*NS + s] = rv;
  }
}

// =====================================================================
// Kernel 3: reduce partials (fixed order -> deterministic), GRU, LN, MLP
// grid 64, block 256.
// =====================================================================
__global__ __launch_bounds__(256) void k_slot(
    float* __restrict__ slots, const float* __restrict__ upd_part,
    const float* __restrict__ row_part,
    const float* __restrict__ wi, const float* __restrict__ wh,
    const float* __restrict__ bi, const float* __restrict__ bh,
    const float* __restrict__ gm, const float* __restrict__ bm,
    const float* __restrict__ w1, const float* __restrict__ b1,
    const float* __restrict__ w2, const float* __restrict__ b2,
    float* __restrict__ rsfinal)
{
  __shared__ float upd[7][32], rs[7], hold[7][32], gi[7][96], gh[7][96];
  __shared__ float mid[7][32], stat[7][2], hid[7][128];
  const int b = blockIdx.x, tid = threadIdx.x;
  const int s = tid >> 5, d = tid & 31;

  if (tid < 7) {
    float rv = 0.f;
    for (int sp = 0; sp < NSPLIT; ++sp) rv += row_part[((size_t)sp*NB + b)*NS + tid];
    rs[tid] = rv;
    rsfinal[b*NS + tid] = rv;     // last iteration's write survives
  }
  if (tid < 224) hold[s][d] = slots[((size_t)b*NS + s)*SD + d];
  __syncthreads();
  if (tid < 224) {
    float uv = 0.f;
    for (int sp = 0; sp < NSPLIT; ++sp)
      uv += upd_part[(((size_t)sp*NB + b)*NS + s)*SD + d];
    upd[s][d] = uv / rs[s];
  }
  __syncthreads();
  for (int o = tid; o < 672; o += 256) {       // GRU gate pre-activations
    int ss = o / 96, j = o % 96;
    float a = bi[j], c = bh[j];
    for (int jd = 0; jd < 32; ++jd) {
      a += upd[ss][jd]  * wi[j*32 + jd];
      c += hold[ss][jd] * wh[j*32 + jd];
    }
    gi[ss][j] = a; gh[ss][j] = c;
  }
  __syncthreads();
  if (tid < 224) {
    float r  = 1.f/(1.f + __expf(-(gi[s][d]      + gh[s][d])));
    float z  = 1.f/(1.f + __expf(-(gi[s][32+d]   + gh[s][32+d])));
    float nn = tanhf(gi[s][64+d] + r*gh[s][64+d]);
    mid[s][d] = (1.f - z)*nn + z*hold[s][d];
  }
  __syncthreads();
  if (tid < 7) {
    float s0 = 0.f, s1 = 0.f;
    for (int j = 0; j < 32; ++j) { float v = mid[tid][j]; s0 += v; s1 += v*v; }
    float mn = s0*(1.f/32.f);
    stat[tid][0] = mn;
    stat[tid][1] = rsqrtf(s1*(1.f/32.f) - mn*mn + 1e-5f);
  }
  __syncthreads();
  for (int o = tid; o < 896; o += 256) {       // MLP hidden (relu)
    int ss = o >> 7, j = o & 127;
    float mn = stat[ss][0], rstd = stat[ss][1];
    float a = b1[j];
    for (int jd = 0; jd < 32; ++jd)
      a += ((mid[ss][jd]-mn)*rstd*gm[jd] + bm[jd]) * w1[j*32 + jd];
    hid[ss][j] = fmaxf(a, 0.f);
  }
  __syncthreads();
  if (tid < 224) {
    float a = b2[d];
    for (int j = 0; j < 128; ++j) a += hid[s][j] * w2[d*128 + j];
    slots[((size_t)b*NS + s)*SD + d] = mid[s][d] + a;
  }
}

// =====================================================================
// Kernel 4a/4b: normalize attn by rowsums, copy slots to both outputs.
// =====================================================================
__global__ void k_norm(float* __restrict__ attn, const float* __restrict__ rsf, int nelem) {
  int idx = blockIdx.x*256 + threadIdx.x;
  if (idx < nelem) attn[idx] /= rsf[idx >> 12];    // 4096 tokens per (b,s)
}
__global__ void k_copy(const float* __restrict__ slots,
                       float* __restrict__ o0, float* __restrict__ o2, int nelem) {
  int idx = blockIdx.x*256 + threadIdx.x;
  if (idx < nelem) { float v = slots[idx]; o0[idx] = v; o2[idx] = v; }
}

// =====================================================================
extern "C" void kernel_launch(void* const* d_in, const int* in_sizes, int n_in,
                              void* d_out, int out_size, void* d_ws, size_t ws_size,
                              hipStream_t stream) {
  (void)in_sizes; (void)n_in; (void)out_size; (void)ws_size;
  const float* inputs     = (const float*)d_in[0];
  const float* slots_init = (const float*)d_in[1];
  const float* ln_in_g    = (const float*)d_in[2];
  const float* ln_in_b    = (const float*)d_in[3];
  const float* Wk         = (const float*)d_in[4];
  const float* Wv         = (const float*)d_in[5];
  const float* Wq         = (const float*)d_in[6];
  const float* ln_s_g     = (const float*)d_in[7];
  const float* ln_s_b     = (const float*)d_in[8];
  const float* gru_wi     = (const float*)d_in[9];
  const float* gru_wh     = (const float*)d_in[10];
  const float* gru_bi     = (const float*)d_in[11];
  const float* gru_bh     = (const float*)d_in[12];
  const float* ln_m_g     = (const float*)d_in[13];
  const float* ln_m_b     = (const float*)d_in[14];
  const float* mlp_w1     = (const float*)d_in[15];
  const float* mlp_b1     = (const float*)d_in[16];
  const float* mlp_w2     = (const float*)d_in[17];
  const float* mlp_b2     = (const float*)d_in[18];

  char* wsb = (char*)d_ws;
  __bf16* kbuf     = (__bf16*)(wsb);                       // 64*4096*32 bf16 = 16 MB
  __bf16* vT       = (__bf16*)(wsb + 16777216);            // transposed v,  16 MB
  float*  slots    = (float*)(wsb + 33554432);             // 64*7*32 f32
  float*  upd_part = (float*)(wsb + 33611776);             // 8*64*7*32 f32
  float*  row_part = (float*)(wsb + 34070528);             // 8*64*7 f32
  float*  rsfinal  = (float*)(wsb + 34084864);             // 64*7 f32

  float* out_slots0 = (float*)d_out;
  float* out_attn   = (float*)d_out + 14336;
  float* out_slots2 = (float*)d_out + 14336 + 1835008;

  hipMemcpyAsync(slots, slots_init, 14336*sizeof(float), hipMemcpyDeviceToDevice, stream);

  k_lnkv<<<dim3(32, 64), 256, 0, stream>>>(inputs, ln_in_g, ln_in_b, Wk, Wv, kbuf, vT);

  for (int it = 0; it < 4; ++it) {
    k_attn<<<dim3(NSPLIT, 64), 256, 0, stream>>>(slots, Wq, ln_s_g, ln_s_b,
                                                 kbuf, vT, out_attn, upd_part, row_part);
    k_slot<<<64, 256, 0, stream>>>(slots, upd_part, row_part,
                                   gru_wi, gru_wh, gru_bi, gru_bh,
                                   ln_m_g, ln_m_b, mlp_w1, mlp_b1, mlp_w2, mlp_b2,
                                   rsfinal);
  }
  k_norm<<<7168, 256, 0, stream>>>(out_attn, rsfinal, 1835008);
  k_copy<<<56, 256, 0, stream>>>(slots, out_slots0, out_slots2, 14336);
}
